// MambaOperator_49933289783629
// MI455X (gfx1250) — matrix operand order repaired and checked by reading, not compiled, not run on hardware
//
#include <hip/hip_runtime.h>

// Mamba block for MI455X (gfx1250, wave32, WMMA).
// Pipeline: f32->bf16 converts, WMMA-bf16 in-proj, fused conv+SiLU,
// WMMA x-proj, WMMA dt-proj (+fused softplus), 2-pass chunked selective scan
// (fused gating, async global->LDS staging), WMMA out-proj.

#define B_SZ    8
#define T_LEN   2048
#define D_MODEL 512
#define D_INNER 1024
#define DT_RANK 32
#define D_STATE 16
#define D_CONV  4
#define MT      (B_SZ * T_LEN)      // 16384 rows of the flattened (b,t) matrix
#define NCHUNK  16
#define CLEN    (T_LEN / NCHUNK)    // 128 steps per scan chunk

typedef __bf16 bf16_t;
typedef __attribute__((ext_vector_type(16))) bf16_t      v16bf;
typedef __attribute__((ext_vector_type(8)))  float       v8f;
typedef __attribute__((ext_vector_type(4)))  unsigned int u32x4;

union FragBF { v16bf v; u32x4 q[2]; };

__device__ __forceinline__ unsigned short f2bf(float f) {
    unsigned int u = __float_as_uint(f);
    unsigned int r = u + 0x7FFFu + ((u >> 16) & 1u);   // round-to-nearest-even
    return (unsigned short)(r >> 16);
}
__device__ __forceinline__ float bf2f(unsigned short s) {
    return __uint_as_float(((unsigned int)s) << 16);
}

// Async global->LDS copy (CDNA5 GLOBAL_LOAD_ASYNC_TO_LDS_B128, ASYNCcnt).
// ldsoff = LDS byte offset (low 32 bits of generic shared pointer), gptr = global.
__device__ __forceinline__ void async_copy_b128_to_lds(unsigned ldsoff,
                                                       const void* gptr) {
    asm volatile("global_load_async_to_lds_b128 %0, %1, off"
                 :: "v"(ldsoff), "v"(gptr) : "memory");
}
__device__ __forceinline__ void wait_asynccnt0() {
    asm volatile("s_wait_asynccnt 0x0" ::: "memory");
}

// ---------------------------------------------------------------- converts
__global__ void f32_to_bf16_kernel(const float* __restrict__ src,
                                   unsigned short* __restrict__ dst, int n) {
    int i = blockIdx.x * blockDim.x + threadIdx.x;
    if (i < n) dst[i] = f2bf(src[i]);
}

// ---------------------------------------------------------------- WMMA GEMM
// C(M,N) = A(M,K) * W(N,K)^T ; A,W bf16, accumulate f32.
// One wave computes one 16x16 tile; K is a compile-time constant so the
// K-loop (steps of 32, v_wmma_f32_16x16x32_bf16) fully unrolls into
// straight-line code: the scheduler hoists fragment loads across earlier
// WMMAs and uses staggered s_wait_loadcnt values — software pipelining
// without manual ping-pong or loop SALU overhead.
// Fragment layouts per CDNA5 ISA 7.12.2:
//  A (16-bit 16x32): lane half (lane>>4) selects K-octet; per lane two 16B loads.
//  B (16-bit 32x16): lane%16 = N column, lane>>4 selects K-octet; two 16B loads, same as A.
//  C/D f32: VGPR r -> M = r + 8*(lane>>4), N = lane%16.
template <int K>
__global__ void wmma_gemm_bf16_kernel(const unsigned short* __restrict__ A,
                                      const unsigned short* __restrict__ Bw,
                                      float* __restrict__ Cf,
                                      unsigned short* __restrict__ Cbf,
                                      const float* __restrict__ bias,
                                      int act, int M, int N) {
    const int lane = threadIdx.x & 31;
    const int wave = threadIdx.x >> 5;
    const int tiles_n = N >> 4;
    const int tile = blockIdx.x * (blockDim.x >> 5) + wave;
    if (tile >= (M >> 4) * tiles_n) return;   // wave-uniform: EXEC stays all-ones
    const int tn = tile % tiles_n;
    const int tm = tile / tiles_n;

    const int half = lane >> 4;
    const int l16  = lane & 15;

    const unsigned short* pa = A  + (size_t)(tm * 16 + l16) * (size_t)K + half * 8;
    const unsigned short* pb = Bw + (size_t)(tn * 16 + l16) * (size_t)K + half * 8;

    v8f acc = {};
#pragma unroll
    for (int k = 0; k < K; k += 32) {
        const unsigned short* qa = pa + k;
        const unsigned short* qb = pb + k;
        FragBF a, b;
        a.q[0] = *(const u32x4*)(qa);
        a.q[1] = *(const u32x4*)(qa + 16);
        b.q[0] = *(const u32x4*)(qb);
        b.q[1] = *(const u32x4*)(qb + 16);
        if (k + 32 < K) {
            __builtin_prefetch(qa + 32, 0, 3);   // -> global_prefetch_b8
            __builtin_prefetch(qb + 32, 0, 3);
        }
        acc = __builtin_amdgcn_wmma_f32_16x16x32_bf16(
                  false, a.v, false, b.v, (short)0, acc, false, false);
    }

    const int n     = tn * 16 + l16;
    const int mbase = tm * 16 + half * 8;
    const float bv  = bias ? bias[n] : 0.0f;
#pragma unroll
    for (int r = 0; r < 8; ++r) {
        float v = acc[r] + bv;
        if (act == 1) {                        // softplus (fused dt activation)
            v = (v > 20.0f) ? v : __logf(1.0f + __expf(v));
        }
        size_t idx = (size_t)(mbase + r) * (size_t)N + n;
        if (Cf)  Cf[idx]  = v;
        if (Cbf) Cbf[idx] = f2bf(v);
    }
}

// ------------------------------------------------- depthwise causal conv + SiLU
__global__ void conv_silu_kernel(const unsigned short* __restrict__ xz_bf,
                                 const float* __restrict__ conv_w,
                                 const float* __restrict__ conv_b,
                                 unsigned short* __restrict__ xc_bf) {
    int idx = blockIdx.x * blockDim.x + threadIdx.x;
    if (idx >= MT * D_INNER) return;
    int d  = idx & (D_INNER - 1);
    int bt = idx >> 10;
    int t  = bt & (T_LEN - 1);
    int b  = bt >> 11;
    float acc = conv_b[d];
#pragma unroll
    for (int k = 0; k < D_CONV; ++k) {
        int ts = t - (D_CONV - 1) + k;
        if (ts >= 0) {
            float xv = bf2f(xz_bf[((size_t)(b * T_LEN + ts)) * (2 * D_INNER) + d]);
            acc += conv_w[d * D_CONV + k] * xv;
        }
    }
    float s = acc / (1.0f + __expf(-acc));     // SiLU
    xc_bf[idx] = f2bf(s);
}

// --------------------------------------------- pull dt_r slice (cols 0..31) as bf16
__global__ void dtr_extract_kernel(const float* __restrict__ dbl,
                                   unsigned short* __restrict__ dtr) {
    int idx = blockIdx.x * blockDim.x + threadIdx.x;
    if (idx >= MT * DT_RANK) return;
    int r = idx >> 5, j = idx & 31;
    dtr[idx] = f2bf(dbl[(size_t)r * 64 + j]);
}

// ---------------------------------------------------------------- scan pass 1
// Per (b, chunk, d): run the recurrence from h=0 over CLEN steps.
// Chunk decay product = exp(A[n] * sum(dt)) so we only need Σdt + final state F.
// B/C columns (shared by all 1024 channels) are staged through LDS with
// async global->LDS copies.
__global__ void scan_pass1_kernel(const float* __restrict__ dt,
                                  const unsigned short* __restrict__ xc_bf,
                                  const float* __restrict__ dbl,
                                  const float* __restrict__ A_log,
                                  float* __restrict__ Fbuf,
                                  float* __restrict__ sumdt) {
    __shared__ float sBC[32][32];              // 32 t-steps x (B[0..15], C[0..15])
    const int tid  = threadIdx.x;
    const int bid  = blockIdx.x;
    const int dblk = bid & 3;
    const int c    = (bid >> 2) & (NCHUNK - 1);
    const int b    = bid >> 6;
    const int d    = dblk * 256 + tid;
    const int t0   = c * CLEN;

    float Av[D_STATE];
#pragma unroll
    for (int n = 0; n < D_STATE; ++n) Av[n] = -__expf(A_log[d * D_STATE + n]);
    float h[D_STATE];
#pragma unroll
    for (int n = 0; n < D_STATE; ++n) h[n] = 0.0f;
    float sdt = 0.0f;

    const int ldl = tid >> 3, ldj = (tid & 7) * 4;
    const unsigned ldsoff = (unsigned)(size_t)&sBC[ldl][ldj];
    for (int tt = 0; tt < CLEN; tt += 32) {
        __syncthreads();
        const float* src = dbl + ((size_t)(b * T_LEN + t0 + tt + ldl)) * 64 + 32 + ldj;
        async_copy_b128_to_lds(ldsoff, src);
        wait_asynccnt0();
        __syncthreads();
        for (int tl = 0; tl < 32; ++tl) {
            int bt = b * T_LEN + t0 + tt + tl;
            float dtv = dt[(size_t)bt * D_INNER + d];
            float xcv = bf2f(xc_bf[(size_t)bt * D_INNER + d]);
            float du  = dtv * xcv;
            sdt += dtv;
#pragma unroll
            for (int n = 0; n < D_STATE; ++n) {
                float a = __expf(dtv * Av[n]);
                h[n] = a * h[n] + du * sBC[tl][n];
            }
        }
    }
    size_t base = ((size_t)((b * NCHUNK + c) * D_INNER + d)) * D_STATE;
#pragma unroll
    for (int n = 0; n < D_STATE; ++n) Fbuf[base + n] = h[n];
    sumdt[(b * NCHUNK + c) * D_INNER + d] = sdt;
}

// ----------------------------------------------------- chunk prefix fix-up (tiny)
__global__ void scan_fix_kernel(const float* __restrict__ Fbuf,
                                const float* __restrict__ sumdt,
                                const float* __restrict__ A_log,
                                float* __restrict__ hin) {
    int idx = blockIdx.x * blockDim.x + threadIdx.x;
    if (idx >= B_SZ * D_INNER) return;
    int b = idx >> 10, d = idx & (D_INNER - 1);
    float Av[D_STATE];
#pragma unroll
    for (int n = 0; n < D_STATE; ++n) Av[n] = -__expf(A_log[d * D_STATE + n]);
    float h[D_STATE];
#pragma unroll
    for (int n = 0; n < D_STATE; ++n) h[n] = 0.0f;
    for (int c = 0; c < NCHUNK; ++c) {
        size_t base = ((size_t)((b * NCHUNK + c) * D_INNER + d)) * D_STATE;
        float sdt = sumdt[(b * NCHUNK + c) * D_INNER + d];
#pragma unroll
        for (int n = 0; n < D_STATE; ++n) {
            hin[base + n] = h[n];
            h[n] = __expf(Av[n] * sdt) * h[n] + Fbuf[base + n];
        }
    }
}

// ---------------------------------------------------------------- scan pass 2
// Replay each chunk from its correct incoming state; emit gated y in bf16.
__global__ void scan_pass2_kernel(const float* __restrict__ dt,
                                  const unsigned short* __restrict__ xc_bf,
                                  const float* __restrict__ dbl,
                                  const float* __restrict__ A_log,
                                  const float* __restrict__ D_skip,
                                  const unsigned short* __restrict__ xz_bf,
                                  const float* __restrict__ hin,
                                  unsigned short* __restrict__ y_bf) {
    __shared__ float sBC[32][32];
    const int tid  = threadIdx.x;
    const int bid  = blockIdx.x;
    const int dblk = bid & 3;
    const int c    = (bid >> 2) & (NCHUNK - 1);
    const int b    = bid >> 6;
    const int d    = dblk * 256 + tid;
    const int t0   = c * CLEN;

    float Av[D_STATE];
#pragma unroll
    for (int n = 0; n < D_STATE; ++n) Av[n] = -__expf(A_log[d * D_STATE + n]);
    float h[D_STATE];
    size_t hbase = ((size_t)((b * NCHUNK + c) * D_INNER + d)) * D_STATE;
#pragma unroll
    for (int n = 0; n < D_STATE; ++n) h[n] = hin[hbase + n];
    const float dsk = D_skip[d];

    const int ldl = tid >> 3, ldj = (tid & 7) * 4;
    const unsigned ldsoff = (unsigned)(size_t)&sBC[ldl][ldj];
    for (int tt = 0; tt < CLEN; tt += 32) {
        __syncthreads();
        const float* src = dbl + ((size_t)(b * T_LEN + t0 + tt + ldl)) * 64 + 32 + ldj;
        async_copy_b128_to_lds(ldsoff, src);
        wait_asynccnt0();
        __syncthreads();
        for (int tl = 0; tl < 32; ++tl) {
            int bt = b * T_LEN + t0 + tt + tl;
            float dtv = dt[(size_t)bt * D_INNER + d];
            float xcv = bf2f(xc_bf[(size_t)bt * D_INNER + d]);
            float du  = dtv * xcv;
            float y   = 0.0f;
#pragma unroll
            for (int n = 0; n < D_STATE; ++n) {
                float a = __expf(dtv * Av[n]);
                h[n] = a * h[n] + du * sBC[tl][n];
                y += h[n] * sBC[tl][16 + n];
            }
            y += dsk * xcv;
            float z = bf2f(xz_bf[(size_t)bt * (2 * D_INNER) + D_INNER + d]);
            y *= z / (1.0f + __expf(-z));      // * silu(z)
            y_bf[(size_t)bt * D_INNER + d] = f2bf(y);
        }
    }
}

// ---------------------------------------------------------------- launcher
extern "C" void kernel_launch(void* const* d_in, const int* in_sizes, int n_in,
                              void* d_out, int out_size, void* d_ws, size_t ws_size,
                              hipStream_t stream) {
    const float* x      = (const float*)d_in[0];
    const float* W_in   = (const float*)d_in[1];
    const float* conv_w = (const float*)d_in[2];
    const float* conv_b = (const float*)d_in[3];
    const float* W_x    = (const float*)d_in[4];
    const float* W_dt   = (const float*)d_in[5];
    const float* b_dt   = (const float*)d_in[6];
    const float* A_log  = (const float*)d_in[7];
    const float* D_skip = (const float*)d_in[8];
    const float* W_out  = (const float*)d_in[9];
    float* out = (float*)d_out;

    char* ws = (char*)d_ws;
    size_t off = 0;
    auto take = [&](size_t bytes) -> void* {
        void* p = ws + off;
        off += (bytes + 255) & ~(size_t)255;
        return p;
    };
    unsigned short* x_bf    = (unsigned short*)take((size_t)MT * D_MODEL * 2);
    unsigned short* Win_bf  = (unsigned short*)take((size_t)2 * D_INNER * D_MODEL * 2);
    unsigned short* Wx_bf   = (unsigned short*)take((size_t)64 * D_INNER * 2);
    unsigned short* Wdt_bf  = (unsigned short*)take((size_t)D_INNER * DT_RANK * 2);
    unsigned short* Wout_bf = (unsigned short*)take((size_t)D_MODEL * D_INNER * 2);
    unsigned short* xz_bf   = (unsigned short*)take((size_t)MT * 2 * D_INNER * 2);
    unsigned short* xc_bf   = (unsigned short*)take((size_t)MT * D_INNER * 2);
    float*          dbl     = (float*)take((size_t)MT * 64 * 4);
    unsigned short* dtr_bf  = (unsigned short*)take((size_t)MT * DT_RANK * 2);
    float*          dtf     = (float*)take((size_t)MT * D_INNER * 4);
    float*          Fbuf    = (float*)take((size_t)B_SZ * NCHUNK * D_INNER * D_STATE * 4);
    float*          sumdt   = (float*)take((size_t)B_SZ * NCHUNK * D_INNER * 4);
    float*          hinbuf  = (float*)take((size_t)B_SZ * NCHUNK * D_INNER * D_STATE * 4);
    unsigned short* y_bf    = (unsigned short*)take((size_t)MT * D_INNER * 2);

    auto cvt = [&](const float* s, unsigned short* dst, int n) {
        f32_to_bf16_kernel<<<(n + 255) / 256, 256, 0, stream>>>(s, dst, n);
    };
    cvt(x,     x_bf,    MT * D_MODEL);
    cvt(W_in,  Win_bf,  2 * D_INNER * D_MODEL);
    cvt(W_x,   Wx_bf,   64 * D_INNER);
    cvt(W_dt,  Wdt_bf,  D_INNER * DT_RANK);
    cvt(W_out, Wout_bf, D_MODEL * D_INNER);

    // 1) xz = x @ W_in^T   (16384 x 2048, K=512)
    {
        int tiles = (MT / 16) * ((2 * D_INNER) / 16);
        wmma_gemm_bf16_kernel<D_MODEL><<<tiles / 4, 128, 0, stream>>>(
            x_bf, Win_bf, nullptr, xz_bf, nullptr, 0, MT, 2 * D_INNER);
    }
    // 2) xc = silu(causal_conv(x_in) + conv_b)
    conv_silu_kernel<<<(MT * D_INNER) / 256, 256, 0, stream>>>(xz_bf, conv_w, conv_b, xc_bf);
    // 3) dbl = xc @ W_x^T  (16384 x 64, K=1024)
    {
        int tiles = (MT / 16) * (64 / 16);
        wmma_gemm_bf16_kernel<D_INNER><<<tiles / 4, 128, 0, stream>>>(
            xc_bf, Wx_bf, dbl, nullptr, nullptr, 0, MT, 64);
    }
    dtr_extract_kernel<<<(MT * DT_RANK) / 256, 256, 0, stream>>>(dbl, dtr_bf);
    // 4) dt = softplus(dt_r @ W_dt^T + b_dt)  (16384 x 1024, K=32)
    {
        int tiles = (MT / 16) * (D_INNER / 16);
        wmma_gemm_bf16_kernel<DT_RANK><<<tiles / 4, 128, 0, stream>>>(
            dtr_bf, Wdt_bf, dtf, nullptr, b_dt, 1, MT, D_INNER);
    }
    // 5) selective scan (2-pass chunked) + gating, fused to bf16 y
    scan_pass1_kernel<<<B_SZ * NCHUNK * 4, 256, 0, stream>>>(dtf, xc_bf, dbl, A_log,
                                                             Fbuf, sumdt);
    scan_fix_kernel<<<(B_SZ * D_INNER) / 256, 256, 0, stream>>>(Fbuf, sumdt, A_log, hinbuf);
    scan_pass2_kernel<<<B_SZ * NCHUNK * 4, 256, 0, stream>>>(dtf, xc_bf, dbl, A_log,
                                                             D_skip, xz_bf, hinbuf, y_bf);
    // 6) out = y @ W_out^T (16384 x 512, K=1024)
    {
        int tiles = (MT / 16) * (D_MODEL / 16);
        wmma_gemm_bf16_kernel<D_INNER><<<tiles / 4, 128, 0, stream>>>(
            y_bf, Wout_bf, out, nullptr, nullptr, 0, MT, D_MODEL);
    }
}
